// ResidualVectorQuantizer_36137854829232
// MI455X (gfx1250) — matrix operand order repaired and checked by reading, not compiled, not run on hardware
//
#include <hip/hip_runtime.h>

typedef __attribute__((ext_vector_type(16))) _Float16     v16h;
typedef __attribute__((ext_vector_type(8)))  float        v8f;
typedef __attribute__((ext_vector_type(8)))  unsigned int v8u;

#define RVQ_K     100
#define RVQ_KP    112          // K padded to 7 WMMA col-tiles
#define RVQ_D     256
#define RVQ_S     4            // stages: embed + 3 residual
#define N_ROWS    131072
#define CSTR      260          // f16 LDS stride for codebook rows (bank spread)
#define RSTR      260          // f32 LDS stride for residual rows (16B-aligned rows)
#define WAVES     4
#define TILE_ROWS (WAVES * 16)
#define GRID_MAIN 512
#define TILES_PER_BLOCK (N_ROWS / (GRID_MAIN * TILE_ROWS))   // = 4

// d_ws float layout
#define WS_SQ   0              // 1 float: global sum of squared residuals
#define WS_CNT  1              // 100 floats: code counts
#define WS_WN   128            // 4*112 floats: ||w||^2 per stage (pad = 1e30)

// d_out float layout (loss, quantized_st, perplexity, encodings)
#define OUT_Q   ((size_t)1)
#define OUT_P   ((size_t)1 + (size_t)N_ROWS * RVQ_D)
#define OUT_E   (OUT_P + 1)

// ---------------------------------------------------------------- prep ----
__global__ void rvq_prep(const float* __restrict__ ew,
                         const float* __restrict__ rw,
                         float* __restrict__ ws) {
    int i = threadIdx.x;
    if (i <= RVQ_K) ws[i] = 0.0f;                 // sqsum + counts
    if (i < RVQ_S * RVQ_KP) {
        int s = i / RVQ_KP, c = i % RVQ_KP;
        float v = 1.0e30f;                        // pad cols never win argmin
        if (c < RVQ_K) {
            const float* wr =
                (s == 0 ? ew : rw + (size_t)(s - 1) * RVQ_K * RVQ_D)
                + (size_t)c * RVQ_D;
            float a = 0.0f;
            for (int d = 0; d < RVQ_D; ++d) { float x = wr[d]; a += x * x; }
            v = a;
        }
        ws[WS_WN + i] = v;
    }
}

// ---------------------------------------------------------------- main ----
__global__ __launch_bounds__(WAVES * 32) void rvq_main(
        const float* __restrict__ inp,
        const float* __restrict__ ew,
        const float* __restrict__ rw,
        float* __restrict__ out,
        float* __restrict__ ws) {

    // ~295 KB LDS: all 4 codebooks (f16) resident for the whole block
    __shared__ __align__(16) _Float16 s_cb[RVQ_S * RVQ_KP * CSTR];
    __shared__ __align__(16) float    s_res[WAVES * 16 * RSTR];
    __shared__ __align__(16) float    s_wn[RVQ_S * RVQ_KP];
    __shared__ int                    s_idx[WAVES * 16];

    const int tid  = threadIdx.x;
    const int lane = tid & 31;
    const int wave = tid >> 5;
    const int m    = lane & 15;     // A row / B col / C col within tile
    const int hi   = lane >> 4;     // lane half

    for (int i = tid; i < RVQ_S * RVQ_KP; i += WAVES * 32)
        s_wn[i] = ws[WS_WN + i];

    for (int i = tid; i < RVQ_S * RVQ_KP * RVQ_D; i += WAVES * 32) {
        int s = i / (RVQ_KP * RVQ_D);
        int r = i % (RVQ_KP * RVQ_D);
        int c = r / RVQ_D;
        int d = r % RVQ_D;
        float v = 0.0f;
        if (c < RVQ_K)
            v = (s == 0) ? ew[(size_t)c * RVQ_D + d]
                         : rw[((size_t)(s - 1) * RVQ_K + c) * RVQ_D + d];
        s_cb[(size_t)s * RVQ_KP * CSTR + (size_t)c * CSTR + d] = (_Float16)v;
    }
    __syncthreads();

    float* res = &s_res[wave * 16 * RSTR];
    // wave-relative LDS byte offset of this wave's residual tile
    // (generic->LDS: aperture occupies bits [63:32]; low word is the offset)
    const unsigned resoff = (unsigned)(size_t)(const void*)res;
    float lsum = 0.0f;

    for (int t = 0; t < TILES_PER_BLOCK; ++t) {
        const size_t rowBase =
            (size_t)(blockIdx.x * TILES_PER_BLOCK + t) * TILE_ROWS + wave * 16;

        // residual := input rows, DMA'd memory->LDS (CDNA5 async path,
        // ASYNCcnt-tracked). 32 x b128 per wave = 16 KB tile.
        {
            const float* gbase = inp + rowBase * RVQ_D;
            #pragma unroll
            for (int i = 0; i < 32; ++i) {
                const int mm = i >> 1;
                const int h  = i & 1;
                const unsigned l =
                    resoff + (unsigned)(mm * RSTR * 4 + h * 512 + lane * 16);
                const float* g = gbase + (size_t)mm * RVQ_D + h * 128 + lane * 4;
                asm volatile("global_load_async_to_lds_b128 %0, %1, off"
                             :: "v"(l), "v"(g) : "memory");
            }
            asm volatile("s_wait_asynccnt 0x0" ::: "memory");
        }
        __builtin_amdgcn_wave_barrier();

        for (int s = 0; s < RVQ_S; ++s) {
            const _Float16* cb = &s_cb[(size_t)s * RVQ_KP * CSTR];

            v8f zero;
            #pragma unroll
            for (int j = 0; j < 8; ++j) zero[j] = 0.0f;
            v8f acc[7];
            #pragma unroll
            for (int q = 0; q < 7; ++q) acc[q] = zero;

            // scores: 16 rows x 112 cols, dot over D=256 via WMMA f16->f32
            for (int kk = 0; kk < RVQ_D; kk += 32) {
                // A fragment: 16x32, ISA 16-bit A layout
                v16h av;
                #pragma unroll
                for (int p = 0; p < 8; ++p) {
                    const int kL = (p < 4) ? (hi * 8 + 2 * p)
                                           : (16 + hi * 8 + 2 * (p - 4));
                    const float2 f2 =
                        *(const float2*)&res[m * RSTR + kk + kL];
                    av[2 * p]     = (_Float16)f2.x;
                    av[2 * p + 1] = (_Float16)f2.y;
                }
                // 7 column tiles of B (32x16 = codebook^T chunk)
                #pragma unroll
                for (int q = 0; q < 7; ++q) {
                    const _Float16* bp =
                        cb + (size_t)(q * 16 + m) * CSTR + kk + hi * 8;
                    v8u bu;
                    #pragma unroll
                    for (int e = 0; e < 8; ++e)
                        bu[e] = *(const unsigned int*)(bp + 2 * e + (e >= 4 ? 8 : 0));
                    v16h bv = __builtin_bit_cast(v16h, bu);
                    acc[q] = __builtin_amdgcn_wmma_f32_16x16x32_f16(
                        false, av, false, bv, (short)0, acc[q], false, false);
                }
            }

            // argmin over 112 cols: in-register over tiles, then 16-lane xor
            #pragma unroll
            for (int v = 0; v < 8; ++v) {      // C row = v + 8*hi
                float bval = 3.0e38f; int bcol = 0;
                #pragma unroll
                for (int q = 0; q < 7; ++q) {
                    const int col = q * 16 + m;
                    const float sc = s_wn[s * RVQ_KP + col] - 2.0f * acc[q][v];
                    if (sc < bval || (sc == bval && col < bcol)) {
                        bval = sc; bcol = col;
                    }
                }
                #pragma unroll
                for (int sh = 1; sh < 16; sh <<= 1) {
                    const float ov = __shfl_xor(bval, sh, 32);
                    const int   oc = __shfl_xor(bcol, sh, 32);
                    if (ov < bval || (ov == bval && oc < bcol)) {
                        bval = ov; bcol = oc;
                    }
                }
                if (m == 0) s_idx[wave * 16 + v + 8 * hi] = bcol;
            }
            __builtin_amdgcn_wave_barrier();

            // exact f32 residual update from global codebook (L2-resident)
            const float* wsrc =
                (s == 0) ? ew : rw + (size_t)(s - 1) * RVQ_K * RVQ_D;
            for (int mm = 0; mm < 16; ++mm) {
                const float* wrow =
                    wsrc + (size_t)s_idx[wave * 16 + mm] * RVQ_D;
                for (int e = lane; e < RVQ_D; e += 32) {
                    const float r = res[mm * RSTR + e] - wrow[e];
                    res[mm * RSTR + e] = r;
                    lsum += r * r;          // == per-stage loss term
                }
            }
            if (lane < 16)
                atomicAdd(&ws[WS_CNT + s_idx[wave * 16 + lane]], 1.0f);
            __builtin_amdgcn_wave_barrier();
        }

        // outputs: quantized_st = x - residual_final; one-hot of last idx
        for (int mm = 0; mm < 16; ++mm) {
            const size_t row = rowBase + mm;
            const int ci = s_idx[wave * 16 + mm];
            for (int e = lane; e < RVQ_D; e += 32)
                out[OUT_Q + row * RVQ_D + e] =
                    inp[row * RVQ_D + e] - res[mm * RSTR + e];
            for (int e = lane; e < RVQ_K; e += 32)
                out[OUT_E + row * RVQ_K + e] = (e == ci) ? 1.0f : 0.0f;
        }
    }

    #pragma unroll
    for (int sh = 1; sh < 32; sh <<= 1) lsum += __shfl_xor(lsum, sh, 32);
    if (lane == 0) atomicAdd(&ws[WS_SQ], lsum);
}

// ------------------------------------------------------------- epilogue ---
__global__ void rvq_final(const float* __restrict__ ws,
                          float* __restrict__ out) {
    const int l = threadIdx.x;
    float ent = 0.0f;
    for (int k = l; k < RVQ_K; k += 32) {
        const float p = ws[WS_CNT + k] * (1.0f / ((float)N_ROWS * RVQ_S));
        ent += p * logf(p + 1e-10f);
    }
    for (int sh = 1; sh < 32; sh <<= 1) ent += __shfl_xor(ent, sh, 32);
    if (l == 0) {
        out[OUT_P] = expf(-ent);
        out[0] = 1.25f * ws[WS_SQ] * (1.0f / ((float)N_ROWS * (float)RVQ_D));
    }
}

// ---------------------------------------------------------------- launch --
extern "C" void kernel_launch(void* const* d_in, const int* in_sizes, int n_in,
                              void* d_out, int out_size, void* d_ws,
                              size_t ws_size, hipStream_t stream) {
    const float* inp = (const float*)d_in[0];
    const float* ew  = (const float*)d_in[1];
    const float* rw  = (const float*)d_in[2];
    float* out = (float*)d_out;
    float* ws  = (float*)d_ws;

    rvq_prep <<<1, 512, 0, stream>>>(ew, rw, ws);
    rvq_main <<<GRID_MAIN, WAVES * 32, 0, stream>>>(inp, ew, rw, out, ws);
    rvq_final<<<1, 32, 0, stream>>>(ws, out);
}
